// Mix6QNet_44616120271022
// MI455X (gfx1250) — matrix operand order repaired and checked by reading, not compiled, not run on hardware
//
#include <hip/hip_runtime.h>
#include <hip/hip_bf16.h>
#include <cstdint>
#include <cstddef>

// ---------------------------------------------------------------------------
// Mix6-style net for MI455X (gfx1250, wave32, WMMA + TDM).
// - All 128-K GEMMs via v_wmma_f32_16x16x32_f16, fp16 activations.
// - Weights staged into LDS with the Tensor Data Mover (tensor_load_to_lds,
//   s_wait_tensorcnt); B-fragments served by ds_load_b128.
// - dconv+c1 fused per layer, r0_w1+r0_w2+fin triple-fused: intermediate
//   tensors live in a per-wave LDS transpose buffer, halving HBM/L2 traffic
//   (~1.8 GB -> ~0.9 GB at 23.3 TB/s).
// - K-loop kept rolled (#pragma unroll 1) to stay under 256 VGPRs.
// ---------------------------------------------------------------------------

typedef __attribute__((ext_vector_type(16))) _Float16 v16h;
typedef __attribute__((ext_vector_type(8)))  _Float16 v8h;
typedef __attribute__((ext_vector_type(8)))  float    v8f;
typedef __attribute__((ext_vector_type(4)))  unsigned int v4u;
typedef __attribute__((ext_vector_type(4)))  int      v4i;
typedef __attribute__((ext_vector_type(8)))  int      v8i;

#define P_TOT  115200      // 512 * 15 * 15 pixels
#define CIN    128
#define HW_    225
#define WIDTH_ 15
#define NBATCH 512

// _DIR_POS -> (dy,dx) = (r-1, c-1) per direction, 3 taps each
__constant__ int c_dy[4][3] = {{ 0,0,0},{-1,0,1},{-1,0,1},{ 1,0,-1}};
__constant__ int c_dx[4][3] = {{-1,0,1},{ 0,0,0},{-1,0,1},{-1,0, 1}};

constexpr float MAXF_C     = 127.0f / 4.0f;                     // 31.75
constexpr float SCALE_FM_C = (4.0f / 16.0f) * 3600.0f / 256.0f; // 3.515625
constexpr float MAXM_C     = 127.0f / SCALE_FM_C;

__device__ __forceinline__ float silu_f(float v) {
    return v / (1.0f + __expf(-v));
}

__device__ __forceinline__ v8f wmma_f16(v16h a, v16h b, v8f c) {
    return __builtin_amdgcn_wmma_f32_16x16x32_f16(false, a, false, b,
                                                  (short)0, c, false, false);
}

// A fragment: lane row base 'row' (128-half row), chunks at kbase+kh*8 and
// kbase+16+kh*8 (ISA 16-bit 16x32 A layout), masked for boundary taps.
__device__ __forceinline__ v16h loadA(const _Float16* row, int kbase, int kh,
                                      _Float16 msk) {
    v8h lo = *(const v8h*)(row + kbase + kh * 8);
    v8h hi = *(const v8h*)(row + kbase + 16 + kh * 8);
    v16h a;
#pragma unroll
    for (int i = 0; i < 8; ++i) { a[i] = lo[i] * msk; a[i + 8] = hi[i] * msk; }
    return a;
}

// B fragment: lane column base pointer, chunks at kh*8 and 16+kh*8 (matches A).
__device__ __forceinline__ v16h loadB(const _Float16* base, int kh) {
    v8h lo = *(const v8h*)(base + kh * 8);
    v8h hi = *(const v8h*)(base + 16 + kh * 8);
    v16h b;
#pragma unroll
    for (int i = 0; i < 8; ++i) { b[i] = lo[i]; b[i + 8] = hi[i]; }
    return b;
}

// ---------------------------------------------------------------------------
// TDM: DMA a [rows][128] fp16 weight tile from global memory into LDS.
// ---------------------------------------------------------------------------
#if __has_builtin(__builtin_amdgcn_tensor_load_to_lds)
#define USE_TDM 1
__device__ __forceinline__ void tdm_load_w(const _Float16* gsrc,
                                           _Float16* lds, int rows) {
    unsigned int lds_off = (unsigned int)(uintptr_t)lds;   // low 32b = LDS offset
    unsigned long long ga = (unsigned long long)(uintptr_t)gsrc;
    v4u g0;
    g0[0] = 1u;                                   // count=1, user mode
    g0[1] = lds_off;                              // lds_addr
    g0[2] = (unsigned int)(ga & 0xffffffffu);     // global_addr[31:0]
    g0[3] = (unsigned int)((ga >> 32) & 0x1ffffffu) | 0x80000000u; // type=2
    v8i g1;
    g1[0] = 0x00010000;                           // data_size=1 (2 bytes)
    g1[1] = (int)((128u & 0xffffu) << 16);        // tensor_dim0 lo16 = 128
    g1[2] = (int)((unsigned)(rows & 0xffff) << 16); // td0 hi=0 | tensor_dim1 lo16
    g1[3] = (int)(128u << 16);                    // td1 hi=0 | tile_dim0 = 128
    g1[4] = rows & 0xffff;                        // tile_dim1 = rows, tile_dim2=0
    g1[5] = 128;                                  // tensor_dim0_stride[31:0]
    g1[6] = 0;
    g1[7] = 0;
    v4i z4 = {0, 0, 0, 0};
#if __clang_major__ >= 23
    v8i z8 = {0, 0, 0, 0, 0, 0, 0, 0};
    __builtin_amdgcn_tensor_load_to_lds(g0, g1, z4, z4, z8, 0);
#else
    __builtin_amdgcn_tensor_load_to_lds(g0, g1, z4, z4, 0);
#endif
}
#else
#define USE_TDM 0
#endif

__device__ __forceinline__ void coop_copy(_Float16* dst, const _Float16* src, int n) {
    for (int i = threadIdx.x * 8; i < n; i += 128 * 8)
        *(v8h*)(dst + i) = *(const v8h*)(src + i);
}

// ---------------------------------------------------------------------------
// Weight fp32 -> fp16 conversion
// ---------------------------------------------------------------------------
__global__ void cvt_f32_f16(const float* __restrict__ src, _Float16* __restrict__ dst, int n) {
    int i = blockIdx.x * blockDim.x + threadIdx.x;
    if (i < n) dst[i] = (_Float16)src[i];
}

// ---------------------------------------------------------------------------
// Input mapping: directional 3-tap conv, 2 -> 128 channels, SiLU, fp16 out.
// Act layout: [dir][pixel][128] fp16, pixel = b*225 + y*15 + x.
// ---------------------------------------------------------------------------
__global__ void map_kernel(const float* __restrict__ x,      // (512,2,15,15)
                           const float* __restrict__ mw,     // (3,128,2)
                           const float* __restrict__ mb,     // (128,)
                           _Float16* __restrict__ act) {
    int idx = blockIdx.x * blockDim.x + threadIdx.x;
    int dir = blockIdx.y;
    int p  = idx >> 7;
    int co = idx & 127;
    if (p >= P_TOT) return;
    int b = p / HW_, pin = p % HW_;
    int py = pin / WIDTH_, px = pin % WIDTH_;
    float s = mb[co];
#pragma unroll
    for (int t = 0; t < 3; ++t) {
        int ys = py + c_dy[dir][t];
        int xs = px + c_dx[dir][t];
        if (ys >= 0 && ys < WIDTH_ && xs >= 0 && xs < WIDTH_) {
            const float* xp = x + ((size_t)b * 2) * HW_ + ys * WIDTH_ + xs;
            s += mw[(t * 128 + co) * 2 + 0] * xp[0]
               + mw[(t * 128 + co) * 2 + 1] * xp[HW_];
        }
    }
    s = silu_f(s);
    act[(size_t)dir * P_TOT * CIN + (size_t)p * CIN + co] = (_Float16)s;
}

// ---------------------------------------------------------------------------
// Fused residual layer: Y = silu(c1 . silu(dirconv(X))) + X
// Wave = 16-pixel tile x 128 couts. dconv temp lives in per-wave LDS
// transpose buffer (never touches memory). LDS DS ops are in-order within a
// wave, so no barrier is needed around the per-wave transpose.
// ---------------------------------------------------------------------------
__global__ __launch_bounds__(128)
void layer_kernel(const _Float16* __restrict__ Xall,   // [4][P][128]
                  const _Float16* __restrict__ Wd,     // [3][128][128] fp16
                  const _Float16* __restrict__ Wc,     // [128][128] fp16
                  const float* __restrict__ bd,        // [128]
                  const float* __restrict__ bc,        // [128]
                  _Float16* __restrict__ Yall) {       // [4][P][128]
    __shared__ _Float16 sWd[3 * 128 * 128];   // 96 KB
    __shared__ _Float16 sWc[128 * 128];       // 32 KB
    __shared__ _Float16 tbuf[4][16 * 128];    // 16 KB, per-wave transpose tile

#if USE_TDM
    if (threadIdx.x < 32) {
        tdm_load_w(Wd, sWd, 3 * 128);
        tdm_load_w(Wc, sWc, 128);
        __builtin_amdgcn_s_wait_tensorcnt(0);
    }
    __syncthreads();
#else
    coop_copy(sWd, Wd, 3 * 128 * 128);
    coop_copy(sWc, Wc, 128 * 128);
    __syncthreads();
#endif

    const int dir  = blockIdx.y;
    const int wave = threadIdx.x >> 5;
    const int lane = threadIdx.x & 31;
    const int p0   = (blockIdx.x * 4 + wave) * 16;
    const int mrow = lane & 15;
    const int kh   = lane >> 4;
    const int nco  = lane & 15;
    const int p    = p0 + mrow;
    const int pin  = p % HW_;
    const int py   = pin / WIDTH_, px = pin % WIDTH_;

    const _Float16* X = Xall + (size_t)dir * P_TOT * CIN;
    _Float16* tb = tbuf[wave];

    // Per-tap source row + validity mask
    const _Float16* tp[3];
    _Float16 msk[3];
#pragma unroll
    for (int t = 0; t < 3; ++t) {
        int dy = c_dy[dir][t], dx = c_dx[dir][t];
        int ys = py + dy, xs = px + dx;
        bool ok = (ys >= 0 && ys < WIDTH_ && xs >= 0 && xs < WIDTH_);
        int psrc = ok ? (p + dy * WIDTH_ + dx) : p;
        tp[t]  = X + (size_t)psrc * CIN;
        msk[t] = ok ? (_Float16)1.0f : (_Float16)0.0f;
    }

    // ---- GEMM1: dirconv, 3 taps, A from global, B from LDS ----------------
    v8f acc[8];
#pragma unroll
    for (int n = 0; n < 8; ++n)
#pragma unroll
        for (int e = 0; e < 8; ++e) acc[n][e] = 0.0f;

#pragma unroll 1
    for (int kc = 0; kc < 4; ++kc) {
        const int kbase = kc * 32;
#pragma unroll
        for (int t = 0; t < 3; ++t) {
            v16h a = loadA(tp[t], kbase, kh, msk[t]);
#pragma unroll
            for (int n = 0; n < 8; ++n) {
                const _Float16* wp =
                    sWd + ((size_t)(t * 128 + n * 16 + nco)) * CIN + kbase;
                acc[n] = wmma_f16(a, loadB(wp, kh), acc[n]);
            }
        }
    }

    // ---- epilogue1: bias+SiLU -> LDS transpose tile [m][ci] ---------------
#pragma unroll
    for (int n = 0; n < 8; ++n) {
        const int ci = n * 16 + nco;
        const float bs = bd[ci];
#pragma unroll
        for (int e = 0; e < 8; ++e) {
            const int m = e + kh * 8;
            tb[m * 128 + ci] = (_Float16)silu_f(acc[n][e] + bs);
        }
    }
    // (same-wave DS RAW; hardware keeps LDS ops in order per wave)

    // ---- GEMM2: pointwise c1, A from LDS transpose tile, B from LDS -------
#pragma unroll
    for (int n = 0; n < 8; ++n)
#pragma unroll
        for (int e = 0; e < 8; ++e) acc[n][e] = 0.0f;

#pragma unroll 1
    for (int kc = 0; kc < 4; ++kc) {
        const int kbase = kc * 32;
        v16h a = loadA(tb + mrow * 128, kbase, kh, (_Float16)1.0f);
#pragma unroll
        for (int n = 0; n < 8; ++n) {
            const _Float16* wp =
                sWc + ((size_t)(n * 16 + nco)) * CIN + kbase;
            acc[n] = wmma_f16(a, loadB(wp, kh), acc[n]);
        }
    }

    // ---- epilogue2: bias+SiLU+residual(X) -> Y ----------------------------
    _Float16* Y = Yall + (size_t)dir * P_TOT * CIN;
#pragma unroll
    for (int n = 0; n < 8; ++n) {
        const int co = n * 16 + nco;
        const float bs = bc[co];
#pragma unroll
        for (int e = 0; e < 8; ++e) {
            const int pe = p0 + e + kh * 8;
            float v = silu_f(acc[n][e] + bs) + (float)X[(size_t)pe * CIN + co];
            Y[(size_t)pe * CIN + co] = (_Float16)v;
        }
    }
}

// ---------------------------------------------------------------------------
// Triple-fused tail: u = silu(W2 silu(W1 x)) + x ; fin = Wf u + bf (f32 out).
// Neither r0 temp ever touches memory; only fin (64ch f32) is written.
// ---------------------------------------------------------------------------
__global__ __launch_bounds__(128)
void r0fin_kernel(const _Float16* __restrict__ Xall,  // [4][P][128]
                  const _Float16* __restrict__ W1,    // [128][128]
                  const _Float16* __restrict__ W2,    // [128][128]
                  const _Float16* __restrict__ Wf,    // [64][128]
                  const float* __restrict__ b1, const float* __restrict__ b2,
                  const float* __restrict__ bf,
                  float* __restrict__ FinAll) {       // [4][P][64] f32
    __shared__ _Float16 sW1[128 * 128];       // 32 KB
    __shared__ _Float16 sW2[128 * 128];       // 32 KB
    __shared__ _Float16 sWf[64 * 128];        // 16 KB
    __shared__ _Float16 tbuf[4][16 * 128];    // 16 KB

#if USE_TDM
    if (threadIdx.x < 32) {
        tdm_load_w(W1, sW1, 128);
        tdm_load_w(W2, sW2, 128);
        tdm_load_w(Wf, sWf, 64);
        __builtin_amdgcn_s_wait_tensorcnt(0);
    }
    __syncthreads();
#else
    coop_copy(sW1, W1, 128 * 128);
    coop_copy(sW2, W2, 128 * 128);
    coop_copy(sWf, Wf, 64 * 128);
    __syncthreads();
#endif

    const int dir  = blockIdx.y;
    const int wave = threadIdx.x >> 5;
    const int lane = threadIdx.x & 31;
    const int p0   = (blockIdx.x * 4 + wave) * 16;
    const int mrow = lane & 15;
    const int kh   = lane >> 4;
    const int nco  = lane & 15;

    const _Float16* X = Xall + (size_t)dir * P_TOT * CIN;
    _Float16* tb = tbuf[wave];

    v8f acc[8];

    // ---- GEMM1: t = silu(W1 x + b1) -> LDS --------------------------------
#pragma unroll
    for (int n = 0; n < 8; ++n)
#pragma unroll
        for (int e = 0; e < 8; ++e) acc[n][e] = 0.0f;
#pragma unroll 1
    for (int kc = 0; kc < 4; ++kc) {
        const int kbase = kc * 32;
        v16h a = loadA(X + (size_t)(p0 + mrow) * CIN, kbase, kh, (_Float16)1.0f);
#pragma unroll
        for (int n = 0; n < 8; ++n)
            acc[n] = wmma_f16(a, loadB(sW1 + ((size_t)(n * 16 + nco)) * CIN
                                       + kbase, kh), acc[n]);
    }
#pragma unroll
    for (int n = 0; n < 8; ++n) {
        const int ci = n * 16 + nco;
        const float bs = b1[ci];
#pragma unroll
        for (int e = 0; e < 8; ++e)
            tb[(e + kh * 8) * 128 + ci] = (_Float16)silu_f(acc[n][e] + bs);
    }

    // ---- GEMM2: u = silu(W2 t + b2) + x -> LDS (overwrite, in-order) ------
#pragma unroll
    for (int n = 0; n < 8; ++n)
#pragma unroll
        for (int e = 0; e < 8; ++e) acc[n][e] = 0.0f;
#pragma unroll 1
    for (int kc = 0; kc < 4; ++kc) {
        const int kbase = kc * 32;
        v16h a = loadA(tb + mrow * 128, kbase, kh, (_Float16)1.0f);
#pragma unroll
        for (int n = 0; n < 8; ++n)
            acc[n] = wmma_f16(a, loadB(sW2 + ((size_t)(n * 16 + nco)) * CIN
                                       + kbase, kh), acc[n]);
    }
#pragma unroll
    for (int n = 0; n < 8; ++n) {
        const int ci = n * 16 + nco;
        const float bs = b2[ci];
#pragma unroll
        for (int e = 0; e < 8; ++e) {
            const int pe = p0 + e + kh * 8;
            float v = silu_f(acc[n][e] + bs) + (float)X[(size_t)pe * CIN + ci];
            tb[(e + kh * 8) * 128 + ci] = (_Float16)v;
        }
    }

    // ---- GEMM3: fin = Wf u + bf (CO=64) -> f32 ----------------------------
    v8f facc[4];
#pragma unroll
    for (int n = 0; n < 4; ++n)
#pragma unroll
        for (int e = 0; e < 8; ++e) facc[n][e] = 0.0f;
#pragma unroll 1
    for (int kc = 0; kc < 4; ++kc) {
        const int kbase = kc * 32;
        v16h a = loadA(tb + mrow * 128, kbase, kh, (_Float16)1.0f);
#pragma unroll
        for (int n = 0; n < 4; ++n)
            facc[n] = wmma_f16(a, loadB(sWf + ((size_t)(n * 16 + nco)) * CIN
                                        + kbase, kh), facc[n]);
    }
    float* Fin = FinAll + (size_t)dir * P_TOT * 64;
#pragma unroll
    for (int n = 0; n < 4; ++n) {
        const int co = n * 16 + nco;
        const float bs = bf[co];
#pragma unroll
        for (int e = 0; e < 8; ++e) {
            const int pe = p0 + e + kh * 8;
            Fin[(size_t)pe * 64 + co] = facc[n][e] + bs;
        }
    }
}

// ---------------------------------------------------------------------------
// feature = sum_d tanh(fin_d/MAXF)*MAXF ; PReLU ; /4   -> f32 [P][64]
// ---------------------------------------------------------------------------
__global__ void combine_kernel(const float* __restrict__ fin,
                               const float* __restrict__ prelu,
                               float* __restrict__ feat) {
    int idx = blockIdx.x * blockDim.x + threadIdx.x;
    if (idx >= P_TOT * 64) return;
    int co = idx & 63;
    float s = 0.0f;
#pragma unroll
    for (int d = 0; d < 4; ++d)
        s += tanhf(fin[(size_t)d * P_TOT * 64 + idx] * (1.0f / MAXF_C)) * MAXF_C;
    s = (s >= 0.0f) ? s : prelu[co] * s;
    feat[idx] = s * 0.25f;
}

// ---------------------------------------------------------------------------
// Policy head
// ---------------------------------------------------------------------------
__global__ void policy_kernel(const float* __restrict__ feat,
                              const float* __restrict__ pdw_w,
                              const float* __restrict__ pdw_b,
                              const float* __restrict__ ppw_w,
                              const float* __restrict__ pscale,
                              float* __restrict__ out) {
    int p = blockIdx.x * blockDim.x + threadIdx.x;
    if (p >= P_TOT) return;
    int b = p / HW_, pin = p % HW_;
    int py = pin / WIDTH_, px = pin % WIDTH_;
    float acc[32];
#pragma unroll
    for (int c = 0; c < 32; ++c) acc[c] = pdw_b[c];
    for (int r = 0; r < 3; ++r)
        for (int cc = 0; cc < 3; ++cc) {
            int ys = py + r - 1, xs = px + cc - 1;
            if (ys < 0 || ys >= WIDTH_ || xs < 0 || xs >= WIDTH_) continue;
            const float* fp = feat + (size_t)(b * HW_ + ys * WIDTH_ + xs) * 64;
#pragma unroll
            for (int c = 0; c < 32; ++c)
                acc[c] += pdw_w[c * 9 + r * 3 + cc] * fp[c];
        }
    float s = 0.0f;
#pragma unroll
    for (int c = 0; c < 32; ++c) {
        float v = acc[c];
        v = (v >= 0.0f) ? v : v * 0.0625f;
        v = fminf(fmaxf(v, -MAXF_C), MAXF_C);
        s += ppw_w[c] * v;
    }
    s = (s >= 0.0f) ? s : s * 0.0625f;
    out[1536 + p] = s * pscale[0];
}

// ---------------------------------------------------------------------------
// Value head
// ---------------------------------------------------------------------------
__global__ __launch_bounds__(32)
void value_kernel(const float* __restrict__ feat,
                  const float* __restrict__ w1, const float* __restrict__ b1,
                  const float* __restrict__ w2, const float* __restrict__ b2,
                  const float* __restrict__ wf, const float* __restrict__ bf,
                  const float* __restrict__ vscale,
                  float* __restrict__ out) {
    __shared__ float sv[32];
    int b = blockIdx.x, c = threadIdx.x;
    float s = 0.0f;
    for (int i = 0; i < HW_; ++i)
        s += feat[(size_t)(b * HW_ + i) * 64 + 32 + c];
    s = fmaxf(s * (1.0f / 225.0f), 0.0f);
    sv[c] = s;
    __syncthreads();
    float h = b1[c];
#pragma unroll
    for (int j = 0; j < 32; ++j) h += w1[c * 32 + j] * sv[j];
    h = fminf(fmaxf(h, 0.0f), MAXM_C);
    __syncthreads(); sv[c] = h; __syncthreads();
    float h2 = b2[c];
#pragma unroll
    for (int j = 0; j < 32; ++j) h2 += w2[c * 32 + j] * sv[j];
    h2 = fminf(fmaxf(h2, 0.0f), MAXM_C);
    __syncthreads(); sv[c] = h2; __syncthreads();
    if (c < 3) {
        float o = bf[c];
#pragma unroll
        for (int j = 0; j < 32; ++j) o += wf[c * 32 + j] * sv[j];
        out[b * 3 + c] = o * vscale[0];
    }
}

// ---------------------------------------------------------------------------
// Orchestration
// ---------------------------------------------------------------------------
extern "C" void kernel_launch(void* const* d_in, const int* in_sizes, int n_in,
                              void* d_out, int out_size, void* d_ws, size_t ws_size,
                              hipStream_t stream) {
    (void)in_sizes; (void)n_in; (void)out_size; (void)ws_size;

    const float* x       = (const float*)d_in[0];
    const float* map_w   = (const float*)d_in[1];
    const float* map_b   = (const float*)d_in[2];
    const float* dconv_w = (const float*)d_in[3];
    const float* dconv_b = (const float*)d_in[4];
    const float* c1_w    = (const float*)d_in[5];
    const float* c1_b    = (const float*)d_in[6];
    const float* r0_w1   = (const float*)d_in[7];
    const float* r0_b1   = (const float*)d_in[8];
    const float* r0_w2   = (const float*)d_in[9];
    const float* r0_b2   = (const float*)d_in[10];
    const float* fin_w   = (const float*)d_in[11];
    const float* fin_b   = (const float*)d_in[12];
    const float* prelu_w = (const float*)d_in[13];
    const float* pdw_w   = (const float*)d_in[14];
    const float* pdw_b   = (const float*)d_in[15];
    const float* ppw_w   = (const float*)d_in[16];
    const float* vl1_w   = (const float*)d_in[17];
    const float* vl1_b   = (const float*)d_in[18];
    const float* vl2_w   = (const float*)d_in[19];
    const float* vl2_b   = (const float*)d_in[20];
    const float* vlf_w   = (const float*)d_in[21];
    const float* vlf_b   = (const float*)d_in[22];
    const float* pscale  = (const float*)d_in[23];
    const float* vscale  = (const float*)d_in[24];
    float* out = (float*)d_out;

    // Workspace: ActA | ActB (fp16 [4][P][128] each; ActB reused as f32 fin),
    // then fp16 weights. Total ~237 MB.
    const size_t ACT = (size_t)4 * P_TOT * CIN * sizeof(_Float16);
    char* ws = (char*)d_ws;
    _Float16* ActA = (_Float16*)ws;
    _Float16* ActB = (_Float16*)(ws + ACT);
    _Float16* Wh   = (_Float16*)(ws + 2 * ACT);
    _Float16* dconvWh = Wh;                       // 4*3*128*128 = 196608
    _Float16* c1Wh    = dconvWh + 196608;         // 65536
    _Float16* r0w1h   = c1Wh + 65536;             // 16384
    _Float16* r0w2h   = r0w1h + 16384;            // 16384
    _Float16* finWh   = r0w2h + 16384;            // 8192

    cvt_f32_f16<<<(196608 + 255) / 256, 256, 0, stream>>>(dconv_w, dconvWh, 196608);
    cvt_f32_f16<<<(65536  + 255) / 256, 256, 0, stream>>>(c1_w,    c1Wh,    65536);
    cvt_f32_f16<<<(16384  + 255) / 256, 256, 0, stream>>>(r0_w1,   r0w1h,   16384);
    cvt_f32_f16<<<(16384  + 255) / 256, 256, 0, stream>>>(r0_w2,   r0w2h,   16384);
    cvt_f32_f16<<<(8192   + 255) / 256, 256, 0, stream>>>(fin_w,   finWh,   8192);

    {
        dim3 g((P_TOT * 128 + 255) / 256, 4);
        map_kernel<<<g, 256, 0, stream>>>(x, map_w, map_b, ActA);
    }

    const dim3 ggemm(P_TOT / 16 / 4, 4);  // 1800 blocks x 4 dirs, 4 waves/block

    // 4 fused residual layers, ping-pong A <-> B
    _Float16* cur = ActA;
    _Float16* nxt = ActB;
    for (int l = 0; l < 4; ++l) {
        layer_kernel<<<ggemm, 128, 0, stream>>>(
            cur, dconvWh + (size_t)l * 3 * 128 * 128, c1Wh + (size_t)l * 128 * 128,
            dconv_b + l * 128, c1_b + l * 128, nxt);
        _Float16* t = cur; cur = nxt; nxt = t;
    }
    // after 4 swaps cur == ActA

    // fused r0 + fin -> f32 fin into the other buffer
    float* finF = (float*)nxt;     // nxt == ActB
    r0fin_kernel<<<ggemm, 128, 0, stream>>>(
        cur, r0w1h, r0w2h, finWh, r0_b1, r0_b2, fin_b, finF);

    // combine -> feature f32 [P][64] in cur (ActA) region
    float* feat = (float*)cur;
    combine_kernel<<<(P_TOT * 64 + 255) / 256, 256, 0, stream>>>(finF, prelu_w, feat);

    policy_kernel<<<(P_TOT + 255) / 256, 256, 0, stream>>>(
        feat, pdw_w, pdw_b, ppw_w, pscale, out);
    value_kernel<<<NBATCH, 32, 0, stream>>>(
        feat, vl1_w, vl1_b, vl2_w, vl2_b, vlf_w, vlf_b, vscale, out);
}
